// GATCurve_32332513804857
// MI455X (gfx1250) — matrix operand order repaired and checked by reading, not compiled, not run on hardware
//
#include <hip/hip_runtime.h>
#include <hip/hip_bf16.h>
#include <stdint.h>

// ---------------------------------------------------------------------------
// GAT with curve (Bezier) parameters, CDNA5 / gfx1250.
// Dense feature transforms: v_wmma_f32_16x16x32_f16 (f32 accum).
//   - A tiles (f16 activations, 16 contiguous rows) staged into LDS with
//     GLOBAL_LOAD_ASYNC_TO_LDS_B128 (+ s_wait_asynccnt), fragments read via
//     ds_load_b128.
//   - B = pre-transposed f16 weights -> contiguous global_load_b128 fragments.
// Attention segment ops use global atomics; edges include self-loops.
// ---------------------------------------------------------------------------

typedef __attribute__((ext_vector_type(16))) _Float16 v16h;
typedef __attribute__((ext_vector_type(8)))  _Float16 v8h;
typedef __attribute__((ext_vector_type(8)))  float    v8f;

#define NEG_SLOPE 0.2f

static inline unsigned cdiv(long long a, long long b) { return (unsigned)((a + b - 1) / b); }

// ---------------------------------------------------------------------------
// float atomic max via CAS (portable, works for negative values / -inf init)
// ---------------------------------------------------------------------------
__device__ inline void atomicMaxF(float* addr, float val) {
  unsigned int* ua = (unsigned int*)addr;
  unsigned int old = *ua;
  while (__uint_as_float(old) < val) {
    unsigned int assumed = old;
    old = atomicCAS(ua, assumed, __float_as_uint(val));
    if (old == assumed) break;
  }
}

// self-loop aware edge fetch: edge_index is [2,E] row-major; loops appended
__device__ inline void edge_sd(const int* __restrict__ ei, int E, int e,
                               int& s, int& d) {
  if (e < E) { s = ei[e]; d = ei[E + e]; }
  else       { s = e - E; d = e - E; }
}

// ---------------------------------------------------------------------------
// Curve interpolation (vectors / biases): out[i] = sum_b c[b]*p[b*n+i]
// ---------------------------------------------------------------------------
__global__ void interp_kernel(const float* __restrict__ p,
                              const float* __restrict__ c,
                              float* __restrict__ out, int n) {
  int i = blockIdx.x * blockDim.x + threadIdx.x;
  if (i >= n) return;
  out[i] = c[0] * p[i] + c[1] * p[n + i] + c[2] * p[2 * n + i];
}

// Curve interpolation for GEMM weights: writes f16 TRANSPOSED copy
// p: (3, K, Ncols) row-major   ->  out_t[col*K + row] (f16, col-major)
__global__ void interp_w_kernel(const float* __restrict__ p,
                                const float* __restrict__ c,
                                _Float16* __restrict__ out_t,
                                int K, int Ncols) {
  int i = blockIdx.x * blockDim.x + threadIdx.x;
  int n = K * Ncols;
  if (i >= n) return;
  float v = c[0] * p[i] + c[1] * p[n + i] + c[2] * p[2 * n + i];
  int r = i / Ncols, col = i % Ncols;
  out_t[(size_t)col * K + r] = (_Float16)v;
}

__global__ void fill_kernel(float* __restrict__ p, float v, long long n) {
  long long i = blockIdx.x * (long long)blockDim.x + threadIdx.x;
  if (i < n) p[i] = v;
}

// ---------------------------------------------------------------------------
// Embedding: h[n,j] = sum_k x[n,k]*W[k,j] + b[j]  (K=7) -> f16 (GEMM-1 A)
// ---------------------------------------------------------------------------
__global__ void emb_kernel(const float* __restrict__ x,
                           const float* __restrict__ W,
                           const float* __restrict__ b,
                           _Float16* __restrict__ hh, int Nn) {
  int i = blockIdx.x * blockDim.x + threadIdx.x;
  if (i >= Nn * 64) return;
  int n = i >> 6, j = i & 63;
  float acc = b[j];
#pragma unroll
  for (int k = 0; k < 7; ++k) acc += x[n * 7 + k] * W[k * 64 + j];
  hh[i] = (_Float16)acc;
}

// ---------------------------------------------------------------------------
// WMMA GEMM: C[M,Nc](f32) = A[M,K](f16 row-major) * B[K,Nc]
//   Bt = f16 transposed weights, layout [Nc][K].
//   Block = 128 threads (4 waves). Block tile: 16 rows x (4*NSUB*16) cols.
//   A tile = 16 contiguous rows = one flat 32*K-byte block -> DMA-staged into
//   LDS with global_load_async_to_lds_b128 (ASYNCcnt), then ds_load_b128
//   fragments. K compile-time -> fully unrolled, NSUB*K/32 v_wmma per wave.
//
// Fragment layouts per CDNA5 ISA 7.12.2 (wave32):
//   A 16x32 f16 : lane&15 = row M ; lanes<16 K {0..7,16..23}, lanes>=16
//                 K {8..15,24..31} (elements 0..7 then 8..15).
//   B 32x16 f16 : lane&15 = col N ; same K split as A.
//   C 16x16 f32 : lane&15 = col N ; VGPR v = row v (+8 for lanes>=16).
// ---------------------------------------------------------------------------
template <int K, int NSUB>
__global__ void gemm_wmma(const _Float16* __restrict__ Ah,
                          const _Float16* __restrict__ Bt,
                          float* __restrict__ C,
                          int M, int Nc) {
  __shared__ __align__(16) _Float16 Ash[16 * K];
  const int mtile = blockIdx.x << 4;

  // --- async DMA stage of the contiguous 16xK f16 A tile into LDS ---------
  {
    unsigned lds_base = (unsigned)(uintptr_t)&Ash[0];
    const char* gbase = (const char*)(Ah + (size_t)mtile * K);
#pragma unroll
    for (int i = threadIdx.x * 16; i < 32 * K; i += 128 * 16) {
      unsigned la = lds_base + (unsigned)i;
      unsigned long long ga = (unsigned long long)(uintptr_t)(gbase + i);
      asm volatile("global_load_async_to_lds_b128 %0, %1, off"
                   :: "v"(la), "v"(ga) : "memory");
    }
    asm volatile("s_wait_asynccnt 0" ::: "memory");
  }
  __syncthreads();

  const int lane = threadIdx.x & 31;
  const int wave = threadIdx.x >> 5;
  const int hi   = lane >> 4;                 // 0: lanes 0-15, 1: lanes 16-31
  const int row  = lane & 15;
  const int kA   = hi ? 8 : 0;                // A fragment K base offset
  const int kB   = hi ? 8 : 0;                // B fragment K base offset
  const int cb0  = (blockIdx.y * 4 + wave) * NSUB * 16;  // first col of wave

  v8f zero = {};
  v8f acc[NSUB];
#pragma unroll
  for (int s = 0; s < NSUB; ++s) acc[s] = zero;

#pragma unroll
  for (int k0 = 0; k0 < K; k0 += 32) {
    const _Float16* ap = &Ash[row * K + k0 + kA];
    v8h alo = *(const v8h*)ap;                // ds_load_b128
    v8h ahi = *(const v8h*)(ap + 16);         // ds_load_b128
    v16h a = __builtin_shufflevector(alo, ahi, 0, 1, 2, 3, 4, 5, 6, 7,
                                               8, 9, 10, 11, 12, 13, 14, 15);
#pragma unroll
    for (int s = 0; s < NSUB; ++s) {
      const _Float16* bp = Bt + (size_t)(cb0 + s * 16 + row) * K + k0 + kB;
      __builtin_prefetch(bp + 32, 0, 0);      // global_prefetch_b8 (gfx1250)
      v8h blo = *(const v8h*)bp;              // global_load_b128
      v8h bhi = *(const v8h*)(bp + 16);       // global_load_b128
      v16h b = __builtin_shufflevector(blo, bhi, 0, 1, 2, 3, 4, 5, 6, 7,
                                                 8, 9, 10, 11, 12, 13, 14, 15);
      acc[s] = __builtin_amdgcn_wmma_f32_16x16x32_f16(
          /*neg_a=*/false, a, /*neg_b=*/false, b,
          /*c_mod=*/(short)0, acc[s], /*reuse_a=*/false, /*reuse_b=*/false);
    }
  }

#pragma unroll
  for (int s = 0; s < NSUB; ++s) {
    float* Cp = C + (size_t)(mtile + (hi ? 8 : 0)) * Nc + cb0 + s * 16 + row;
#pragma unroll
    for (int v = 0; v < 8; ++v) Cp[(size_t)v * Nc] = acc[s][v];
  }
}

// ---------------------------------------------------------------------------
// Attention scores: e_src[n,h]=dot(h[n,h,:],a_src[h,:]), same for e_dst
// ---------------------------------------------------------------------------
__global__ void escore_kernel(const float* __restrict__ h,
                              const float* __restrict__ asrc,
                              const float* __restrict__ adst,
                              float* __restrict__ esrc,
                              float* __restrict__ edst,
                              int Nn, int H, int D) {
  int i = blockIdx.x * blockDim.x + threadIdx.x;
  if (i >= Nn * H) return;
  int n = i / H, hd = i % H;
  const float* hp = h + (size_t)n * H * D + (size_t)hd * D;
  float s = 0.f, d = 0.f;
  for (int k = 0; k < D; ++k) {
    float v = hp[k];
    s += v * asrc[hd * D + k];
    d += v * adst[hd * D + k];
  }
  esrc[i] = s;
  edst[i] = d;
}

__global__ void att_max_kernel(const int* __restrict__ ei,
                               const float* __restrict__ esrc,
                               const float* __restrict__ edst,
                               float* __restrict__ amax,
                               int E, int Nn, int H) {
  long long i = blockIdx.x * (long long)blockDim.x + threadIdx.x;
  long long tot = (long long)(E + Nn) * H;
  if (i >= tot) return;
  int e = (int)(i / H), hd = (int)(i % H);
  int s, d; edge_sd(ei, E, e, s, d);
  float a = esrc[(size_t)s * H + hd] + edst[(size_t)d * H + hd];
  a = a > 0.f ? a : a * NEG_SLOPE;
  atomicMaxF(&amax[(size_t)d * H + hd], a);
}

__global__ void att_exp_kernel(const int* __restrict__ ei,
                               const float* __restrict__ esrc,
                               const float* __restrict__ edst,
                               const float* __restrict__ amax,
                               float* __restrict__ alpha,
                               float* __restrict__ denom,
                               int E, int Nn, int H) {
  long long i = blockIdx.x * (long long)blockDim.x + threadIdx.x;
  long long tot = (long long)(E + Nn) * H;
  if (i >= tot) return;
  int e = (int)(i / H), hd = (int)(i % H);
  int s, d; edge_sd(ei, E, e, s, d);
  float a = esrc[(size_t)s * H + hd] + edst[(size_t)d * H + hd];
  a = a > 0.f ? a : a * NEG_SLOPE;
  float ex = __expf(a - amax[(size_t)d * H + hd]);
  alpha[i] = ex;
  atomicAdd(&denom[(size_t)d * H + hd], ex);
}

// one block per edge; blockDim = H*D (256 or 64)
__global__ void att_agg_kernel(const int* __restrict__ ei,
                               const float* __restrict__ h,
                               const float* __restrict__ alpha,
                               const float* __restrict__ denom,
                               float* __restrict__ out,
                               int E, int Nn, int H, int D) {
  int e = blockIdx.x;
  int t = threadIdx.x;            // feature index in [0, H*D)
  int s, d; edge_sd(ei, E, e, s, d);
  int hd = t / D;
  float a = alpha[(size_t)e * H + hd] / denom[(size_t)d * H + hd];
  atomicAdd(&out[(size_t)d * H * D + t], h[(size_t)s * H * D + t] * a);
}

// bias + ELU; optional f16 twin for the next GEMM's A matrix
__global__ void bias_elu_kernel(float* __restrict__ h,
                                _Float16* __restrict__ hh,
                                const float* __restrict__ b,
                                long long n, int F) {
  long long i = blockIdx.x * (long long)blockDim.x + threadIdx.x;
  if (i >= n) return;
  float v = h[i] + b[(int)(i % F)];
  v = v > 0.f ? v : (__expf(v) - 1.f);
  h[i] = v;
  if (hh) hh[i] = (_Float16)v;
}

__global__ void bias_relu_kernel(float* __restrict__ h,
                                 const float* __restrict__ b,
                                 long long n, int F) {
  long long i = blockIdx.x * (long long)blockDim.x + threadIdx.x;
  if (i >= n) return;
  float v = h[i] + b[(int)(i % F)];
  h[i] = v > 0.f ? v : 0.f;
}

// ---------------------------------------------------------------------------
// Graph pooling (mean + max over batch)
// ---------------------------------------------------------------------------
__global__ void pool_kernel(const float* __restrict__ h,
                            const int* __restrict__ batch,
                            float* __restrict__ psum,
                            float* __restrict__ pmax,
                            float* __restrict__ pcnt, int Nn) {
  int i = blockIdx.x * blockDim.x + threadIdx.x;
  if (i >= Nn * 64) return;
  int n = i >> 6, f = i & 63;
  int g = batch[n];
  float v = h[i];
  atomicAdd(&psum[g * 64 + f], v);
  atomicMaxF(&pmax[g * 64 + f], v);
  if (f == 0) atomicAdd(&pcnt[g], 1.f);
}

// -> f16 pooled features [G,128] (A matrix of fc1 WMMA GEMM)
__global__ void pool_finalize(const float* __restrict__ psum,
                              const float* __restrict__ pmax,
                              const float* __restrict__ pcnt,
                              _Float16* __restrict__ g, int G) {
  int i = blockIdx.x * blockDim.x + threadIdx.x;
  if (i >= G * 128) return;
  int gi = i >> 7, f = i & 127;
  float c = pcnt[gi];
  float v;
  if (f < 64) v = psum[gi * 64 + f] / fmaxf(c, 1.f);
  else        v = (c > 0.f) ? pmax[gi * 64 + f - 64] : 0.f;
  g[i] = (_Float16)v;
}

__global__ void fc2_kernel(const float* __restrict__ a,
                           const float* __restrict__ W,
                           const float* __restrict__ b,
                           float* __restrict__ out, int G) {
  int i = blockIdx.x * blockDim.x + threadIdx.x;
  if (i >= G * 10) return;
  int gi = i / 10, j = i % 10;
  float acc = b[j];
  for (int k = 0; k < 64; ++k) acc += a[gi * 64 + k] * W[k * 10 + j];
  out[i] = acc;
}

// ---------------------------------------------------------------------------
extern "C" void kernel_launch(void* const* d_in, const int* in_sizes, int n_in,
                              void* d_out, int out_size, void* d_ws, size_t ws_size,
                              hipStream_t stream) {
  const float* x      = (const float*)d_in[0];
  const int*   ei     = (const int*)d_in[1];      // [2,E]
  const int*   batch  = (const int*)d_in[2];
  const float* coeffs = (const float*)d_in[3];    // [3]
  const float* P[18];
  for (int t = 0; t < 18; ++t) P[t] = (const float*)d_in[4 + t];
  // P order: emb_W, emb_b, g1_W, g1_asrc, g1_adst, g1_b,
  //          g2_W, g2_asrc, g2_adst, g2_b, g3_W, g3_asrc, g3_adst, g3_b,
  //          fc1_W, fc1_b, fc2_W, fc2_b

  const int Nn = in_sizes[0] / 7;
  const int E  = in_sizes[1] / 2;
  const int G  = out_size / 10;
  const int E2 = E + Nn;

  // ---------------- workspace bump allocator ----------------
  char* base = (char*)d_ws;
  size_t off = 0;
  auto alloc = [&](size_t bytes) -> char* {
    char* p = base + off;
    off = (off + bytes + 255) & ~(size_t)255;
    return p;
  };
  auto af = [&](size_t n) { return (float*)alloc(n * 4); };
  auto ah = [&](size_t n) { return (_Float16*)alloc(n * 2); };

  // interpolated f32 params (vectors/biases + small mats)
  float* W_emb = af(448);   float* b_emb = af(64);
  float* a1s = af(256); float* a1d = af(256); float* b1 = af(256);
  float* a2s = af(256); float* a2d = af(256); float* b2 = af(256);
  float* a3s = af(64);  float* a3d = af(64);  float* b3 = af(64);
  float* bf1 = af(64);
  float* Wf2 = af(640); float* bf2 = af(16);
  // f16 transposed weights for WMMA ( [Nc][K] )
  _Float16* W1t  = ah(16384);   // 64  -> 256
  _Float16* W2t  = ah(65536);   // 256 -> 256
  _Float16* W3t  = ah(16384);   // 256 -> 64
  _Float16* Wf1t = ah(8192);    // 128 -> 64
  // activations
  _Float16* h0h  = ah((size_t)Nn * 64);     // emb out (GEMM-1 A)
  float*    hA   = af((size_t)Nn * 256);    // GEMM out / attention in
  float*    hB   = af((size_t)Nn * 256);    // attention out (f32)
  _Float16* hBh  = ah((size_t)Nn * 256);    // attention out (f16, next A)
  float*    h3   = af((size_t)Nn * 64);     // GEMM-3 out
  float*    h3o  = af((size_t)Nn * 64);     // attention-3 out
  float* esrc = af((size_t)Nn * 4);
  float* edst = af((size_t)Nn * 4);
  float* amax = af((size_t)Nn * 4);
  float* denm = af((size_t)Nn * 4);
  float* alph = af((size_t)E2 * 4);
  float* psum = af((size_t)G * 64);
  float* pmax = af((size_t)G * 64);
  float* pcnt = af((size_t)G);
  _Float16* gfeath = ah((size_t)G * 128);   // pooled feats (fc1 A)
  float* f1o  = af((size_t)G * 64);
  (void)ws_size; (void)n_in;

  const float NINF = -__builtin_inff();

  auto interp = [&](const float* p, float* o, int n) {
    interp_kernel<<<cdiv(n, 256), 256, 0, stream>>>(p, coeffs, o, n);
  };
  auto interp_w = [&](const float* p, _Float16* ot, int K, int Ncols) {
    interp_w_kernel<<<cdiv((long long)K * Ncols, 256), 256, 0, stream>>>(
        p, coeffs, ot, K, Ncols);
  };
  interp(P[0],  W_emb, 448);
  interp(P[1],  b_emb, 64);
  interp_w(P[2],  W1t, 64, 256);
  interp(P[3],  a1s, 256);
  interp(P[4],  a1d, 256);
  interp(P[5],  b1,  256);
  interp_w(P[6],  W2t, 256, 256);
  interp(P[7],  a2s, 256);
  interp(P[8],  a2d, 256);
  interp(P[9],  b2,  256);
  interp_w(P[10], W3t, 256, 64);
  interp(P[11], a3s, 64);
  interp(P[12], a3d, 64);
  interp(P[13], b3,  64);
  interp_w(P[14], Wf1t, 128, 64);
  interp(P[15], bf1, 64);
  interp(P[16], Wf2, 640);
  interp(P[17], bf2, 10);

  // embedding: h0h = f16( x @ Wemb + bemb )
  emb_kernel<<<cdiv((long long)Nn * 64, 256), 256, 0, stream>>>(x, W_emb, b_emb, h0h, Nn);

  // attention block helper (h = transformed features [Nn, H*D], out separate)
  auto attention = [&](const float* h, const float* as, const float* ad,
                       float* out, _Float16* outh, const float* bias,
                       int H, int D) {
    long long NH = (long long)Nn * H, EH = (long long)E2 * H;
    long long ND = (long long)Nn * H * D;
    escore_kernel<<<cdiv(NH, 256), 256, 0, stream>>>(h, as, ad, esrc, edst, Nn, H, D);
    fill_kernel<<<cdiv(NH, 256), 256, 0, stream>>>(amax, NINF, NH);
    fill_kernel<<<cdiv(NH, 256), 256, 0, stream>>>(denm, 0.f, NH);
    att_max_kernel<<<cdiv(EH, 256), 256, 0, stream>>>(ei, esrc, edst, amax, E, Nn, H);
    att_exp_kernel<<<cdiv(EH, 256), 256, 0, stream>>>(ei, esrc, edst, amax, alph, denm, E, Nn, H);
    fill_kernel<<<cdiv(ND, 256), 256, 0, stream>>>(out, 0.f, ND);
    att_agg_kernel<<<E2, H * D, 0, stream>>>(ei, h, alph, denm, out, E, Nn, H, D);
    bias_elu_kernel<<<cdiv(ND, 256), 256, 0, stream>>>(out, outh, bias, ND, H * D);
  };

  // ---- GAT layer 1: h0h[Nn,64] @ W1[64,256] -> hA ; attention -> hB/hBh ----
  gemm_wmma<64, 4><<<dim3(Nn / 16, 1), 128, 0, stream>>>(h0h, W1t, hA, Nn, 256);
  attention(hA, a1s, a1d, hB, hBh, b1, 4, 64);

  // ---- GAT layer 2: hBh @ W2[256,256] -> hA ; attention -> hB/hBh ----
  gemm_wmma<256, 4><<<dim3(Nn / 16, 1), 128, 0, stream>>>(hBh, W2t, hA, Nn, 256);
  attention(hA, a2s, a2d, hB, hBh, b2, 4, 64);

  // ---- GAT layer 3: hBh @ W3[256,64] -> h3 ; attention (H=1) -> h3o ----
  gemm_wmma<256, 1><<<dim3(Nn / 16, 1), 128, 0, stream>>>(hBh, W3t, h3, Nn, 64);
  attention(h3, a3s, a3d, h3o, nullptr, b3, 1, 64);

  // ---- pooling: mean | max over graphs -> f16 [G,128] ----
  fill_kernel<<<cdiv((long long)G * 64, 256), 256, 0, stream>>>(psum, 0.f, (long long)G * 64);
  fill_kernel<<<cdiv((long long)G * 64, 256), 256, 0, stream>>>(pmax, NINF, (long long)G * 64);
  fill_kernel<<<cdiv(G, 256), 256, 0, stream>>>(pcnt, 0.f, G);
  pool_kernel<<<cdiv((long long)Nn * 64, 256), 256, 0, stream>>>(h3o, batch, psum, pmax, pcnt, Nn);
  pool_finalize<<<cdiv((long long)G * 128, 256), 256, 0, stream>>>(psum, pmax, pcnt, gfeath, G);

  // ---- fc1 (WMMA): gfeath[G,128] @ Wf1[128,64] -> f1o ; relu ----
  gemm_wmma<128, 1><<<dim3(G / 16, 1), 128, 0, stream>>>(gfeath, Wf1t, f1o, G, 64);
  bias_relu_kernel<<<cdiv((long long)G * 64, 256), 256, 0, stream>>>(f1o, bf1, (long long)G * 64, 64);

  // ---- fc2 (scalar, N=10) -> d_out ----
  fc2_kernel<<<cdiv((long long)G * 10, 256), 256, 0, stream>>>(f1o, Wf2, bf2, (float*)d_out, G);
}
